// EnhancedSpectralExtractor_62508954026588
// MI455X (gfx1250) — matrix operand order repaired and checked by reading, not compiled, not run on hardware
//
#include <hip/hip_runtime.h>
#include <hip/hip_bf16.h>

typedef __attribute__((ext_vector_type(16))) _Float16     v16h;
typedef __attribute__((ext_vector_type(8)))  float        v8f;
typedef __attribute__((ext_vector_type(4)))  unsigned int v4u;
typedef __attribute__((ext_vector_type(8)))  int          v8i;
typedef __attribute__((ext_vector_type(4)))  int          v4i;

#define NTHREADS 256
#define NSAMP    8          // samples (spatial positions) per workgroup
#define SSTRIDE  1152       // per-sample stride (halfs) in activation buffers
#define NCH      143        // spectral channels
#define XCSTRIDE 64         // x stride per channel  (8*8 floats)
#define XBSTRIDE 9152       // x stride per batch    (143*64 floats)

// d_ws layout (halfs): [0,46080) DFT twiddle A-fragments,
//                      [46080,...) packed conv weight A-fragments (w2:4, w3:6, w4:6 frags)
#define TW_HALFS  46080
#define WT2_OFF   0
#define WT3_OFF   2048
#define WT4_OFF   5120

// --- WMMA fragment element -> K mapping (16-bit operands, wave32) -----------
// A-matrix 16x32 (ISA 7.12.2): lanes 0-15 VGPR0..3 hold K=0..7, VGPR4..7 K=16..23;
// lanes 16-31 hold K=8..15 / K=24..31.
__device__ __forceinline__ int kmapA(int e, int g) { return ((e < 8) ? 0 : 16) + g * 8 + (e & 7); }
// B-matrix 32x16: same K order as A: lanes 0-15 hold K=0..7 / K=16..23, lanes 16-31 K=8..15 / K=24..31.
__device__ __forceinline__ int kmapB(int e, int g) { return ((e < 8) ? 0 : 16) + g * 8 + (e & 7); }

// ---------------------------------------------------------------------------
// Init kernel 1: bake 143-point DFT cos/sin matrices (zero-padded to 144x160)
// directly into WMMA A-fragment layout:  tw[trig][tile(9)][ks(5)][lane(32)][e(16)]
// ---------------------------------------------------------------------------
__global__ void dft_twiddle_init(_Float16* __restrict__ tw) {
  int idx  = blockIdx.x;            // [0,90) = trig*45 + tile*5 + ks
  int trig = idx / 45;
  int rem  = idx - trig * 45;
  int tile = rem / 5;
  int ks   = rem - tile * 5;
  int lane = threadIdx.x;
  int g = lane >> 4, m = lane & 15;
  int k = tile * 16 + m;            // frequency index (M dimension)
  _Float16* dst = tw + (size_t)(idx * 32 + lane) * 16;
#pragma unroll
  for (int e = 0; e < 16; ++e) {
    int c = ks * 32 + kmapA(e, g);  // channel index (K dimension)
    float v = 0.f;
    if (c < NCH && k < NCH) {
      float ang = -6.28318530717958647692f * (float)((k * c) % NCH) / (float)NCH;
      v = trig ? __sinf(ang) : __cosf(ang);
    }
    dst[e] = (_Float16)v;
  }
}

// ---------------------------------------------------------------------------
// Init kernel 2: pack conv weights (w2/w3/w4) into WMMA A-fragment layout
// fragment order per block: [mt][ks][lane][e]
// ---------------------------------------------------------------------------
__global__ void weight_pack_init(const float* __restrict__ w2, const float* __restrict__ w3,
                                 const float* __restrict__ w4, _Float16* __restrict__ wt) {
  int idx = blockIdx.x;                 // [0,16)
  const float* w;
  int Ktot, mt, ks, frag, off;
  if (idx < 4)       { w = w2; Ktot = 48; frag = idx;      mt = frag / 2; ks = frag - mt * 2; off = WT2_OFF; }
  else if (idx < 10) { w = w3; Ktot = 84; frag = idx - 4;  mt = frag / 3; ks = frag - mt * 3; off = WT3_OFF; }
  else               { w = w4; Ktot = 84; frag = idx - 10; mt = frag / 3; ks = frag - mt * 3; off = WT4_OFF; }
  const int Cout = 28;
  int lane = threadIdx.x;
  int g = lane >> 4, ln16 = lane & 15;
  int oc = mt * 16 + ln16;
  _Float16* dst = wt + off + (size_t)(frag * 32 + lane) * 16;
#pragma unroll
  for (int e = 0; e < 16; ++e) {
    int K = ks * 32 + kmapA(e, g);
    dst[e] = (oc < Cout && K < Ktot) ? (_Float16)w[oc * Ktot + K] : (_Float16)0.f;
  }
}

// ---------------------------------------------------------------------------
// Small VALU conv (block 1 and the 4-channel FFT branch): out = conv1d(in), stride 2
// ---------------------------------------------------------------------------
__device__ void conv_valu(const _Float16* in, int s_stride, int ic_stride,
                          int Cin, int Cout, int Lout,
                          const float* __restrict__ w,
                          _Float16* out, int tid) {
  int total = NSAMP * Cout * Lout;
  for (int i = tid; i < total; i += NTHREADS) {
    int s  = i / (Cout * Lout);
    int r  = i - s * (Cout * Lout);
    int oc = r / Lout;
    int t  = r - oc * Lout;
    float acc = 0.f;
    for (int ic = 0; ic < Cin; ++ic) {
      const _Float16* src = in + s * s_stride + ic * ic_stride + 2 * t;
      const float*    wr  = w + (oc * Cin + ic) * 3;
      acc += (float)src[0] * wr[0] + (float)src[1] * wr[1] + (float)src[2] * wr[2];
    }
    out[s * SSTRIDE + oc * Lout + t] = (_Float16)acc;
  }
}

// ---------------------------------------------------------------------------
// LayerNorm over the time axis (per sample, per channel) + ReLU, in place
// ---------------------------------------------------------------------------
__device__ void ln_relu(_Float16* buf, int Cout, int Lout,
                        const float* __restrict__ gam,
                        const float* __restrict__ bet, int tid) {
  for (int i = tid; i < NSAMP * Cout; i += NTHREADS) {
    int s = i / Cout, oc = i - s * Cout;
    _Float16* row = buf + s * SSTRIDE + oc * Lout;
    float mu = 0.f;
    for (int t = 0; t < Lout; ++t) mu += (float)row[t];
    mu /= (float)Lout;
    float var = 0.f;
    for (int t = 0; t < Lout; ++t) { float d = (float)row[t] - mu; var += d * d; }
    var /= (float)Lout;
    float inv = rsqrtf(var + 1e-5f);
    for (int t = 0; t < Lout; ++t) {
      float y = ((float)row[t] - mu) * inv * gam[t] + bet[t];
      row[t] = (_Float16)fmaxf(y, 0.f);
    }
  }
}

// ---------------------------------------------------------------------------
// im2col conv via v_wmma_f32_16x16x32_f16.
// A = pre-packed weight fragments (M = OC padded to 32 -> 2 tiles)
// B = gathered activations (N = 8 samples * Lout time positions)
// ---------------------------------------------------------------------------
template<int Cin, int Lin, int Cout, int Lout>
__device__ void conv_wmma(const _Float16* in, const _Float16* __restrict__ wtab,
                          _Float16* out, int tid) {
  const int lane = tid & 31, wave = tid >> 5;
  const int g = lane >> 4, ln16 = lane & 15;
  constexpr int Ktot   = Cin * 3;
  constexpr int ksteps = (Ktot + 31) / 32;
  constexpr int NC     = NSAMP * Lout;
  constexpr int ntiles = (NC + 15) / 16;

  for (int mt = 0; mt < 2; ++mt) {
    v16h afrag[ksteps];
#pragma unroll
    for (int ks = 0; ks < ksteps; ++ks)
      afrag[ks] = *(const v16h*)(wtab + (size_t)(((mt * ksteps + ks) * 32) + lane) * 16);
    for (int nt = wave; nt < ntiles; nt += 8) {  // wave-uniform: EXEC stays full at wmma
      int  cidx  = nt * 16 + ln16;
      bool colok = cidx < NC;
      int  cid   = colok ? cidx : 0;
      int  s     = cid / Lout;
      int  t     = cid - s * Lout;
      v8f acc = {};
#pragma unroll
      for (int ks = 0; ks < ksteps; ++ks) {
        v16h bfrag;
#pragma unroll
        for (int e = 0; e < 16; ++e) {
          int K = ks * 32 + kmapB(e, g);
          _Float16 v = (_Float16)0.f;
          if (colok && K < Ktot) {
            int ic = K / 3;
            int kk = K - ic * 3;
            v = in[s * SSTRIDE + ic * Lin + 2 * t + kk];   // im2col gather from LDS
          }
          bfrag[e] = v;
        }
        acc = __builtin_amdgcn_wmma_f32_16x16x32_f16(false, afrag[ks], false, bfrag,
                                                     (short)0, acc, false, false);
      }
#pragma unroll
      for (int r = 0; r < 8; ++r) {                // D layout: M = r + 8*g, N = ln16
        int oc = mt * 16 + r + g * 8;
        if (colok && oc < Cout)
          out[s * SSTRIDE + oc * Lout + t] = (_Float16)acc[r];
      }
    }
  }
}

// ---------------------------------------------------------------------------
// Fully fused pipeline: one workgroup = one (batch, row/col-stream) pair.
// TDM tensor gather -> WMMA DFT magnitude -> conv/LN/ReLU chain (WMMA blocks 2-4)
// -> FFT-branch chain (VALU) -> stores.  All intermediates live in LDS.
// ---------------------------------------------------------------------------
__global__ __launch_bounds__(NTHREADS) void spectral_pipeline(
    const float* __restrict__ x,
    const float* __restrict__ w1, const float* __restrict__ f1,
    const float* __restrict__ f2, const float* __restrict__ f3,
    const float* __restrict__ f4,
    const float* __restrict__ g1, const float* __restrict__ b1,
    const float* __restrict__ g2, const float* __restrict__ b2,
    const float* __restrict__ g3, const float* __restrict__ b3,
    const float* __restrict__ g4, const float* __restrict__ b4,
    const _Float16* __restrict__ tw, const _Float16* __restrict__ wt,
    float* __restrict__ out) {
  __shared__ float    stage[NSAMP * NCH * 2];  // TDM landing buffer, [s][c][p] f32
  __shared__ _Float16 vmain[NSAMP][2][160];    // raw centre-row/col vectors (zero padded)
  __shared__ _Float16 vfft [NSAMP][2][160];    // DFT magnitudes
  __shared__ _Float16 bufA[NSAMP * SSTRIDE];   // ping
  __shared__ _Float16 bufB[NSAMP * SSTRIDE];   // pong

  const int tid  = threadIdx.x;
  const int lane = tid & 31;
  const int wave = tid >> 5;
  const int g    = lane >> 4;
  const int ln16 = lane & 15;
  const int b    = blockIdx.x;
  const int kind = blockIdx.y;    // 0 = centre rows (cr), 1 = centre cols (cc)

  // -- stage 1: Tensor Data Mover gathers the strided tile into LDS ----------
  // tile: (2 contiguous f32) x (143 rows, stride 64) x (8 samples, stride 2|8)
  if (wave == 0) {
    const int base    = (kind == 0) ? 24 : 3;   // cr: (3,2q+p) -> 24+2s+p ; cc: (h,3+p) -> 8s+3+p
    const int strideS = (kind == 0) ? 2 : 8;
    unsigned long long gaddr = (unsigned long long)(const void*)(x + b * XBSTRIDE + base);
    unsigned int       laddr = (unsigned int)(unsigned long long)(const void*)&stage[0];

    v4u g0;
    g0[0] = 1u;                                          // count=1, user descriptor
    g0[1] = laddr;                                       // lds_addr
    g0[2] = (unsigned int)gaddr;                         // global_addr[31:0]
    g0[3] = ((unsigned int)(gaddr >> 32) & 0x01FFFFFFu)  // global_addr[56:32]
            | (2u << 30);                                // type=2 ("image")
    v8i g1;
    g1[0] = 0x20000;                 // workgroup_mask=0, data_size=2 (4B)
    g1[1] = (int)(2u << 16);         // tensor_dim0 = 2
    g1[2] = (int)(143u << 16);       // tensor_dim1 = 143 (low half in [31:16])
    g1[3] = (int)(2u << 16);         // tile_dim0 = 2
    g1[4] = (int)(143u | (8u << 16));// tile_dim1 = 143, tile_dim2 = 8
    g1[5] = 64;                      // tensor_dim0_stride = 64 elements
    g1[6] = (int)((unsigned)strideS << 16); // tensor_dim1_stride[15:0]
    g1[7] = 0;
    v4i g2;
    g2[0] = 8;                       // tensor_dim2 = 8
    g2[1] = 0; g2[2] = 0; g2[3] = 0; // no dim3, iterate disabled
    v4i g3;
    g3[0] = 0; g3[1] = 0; g3[2] = 0; g3[3] = 0;
    v8i g4;                          // extra descriptor group (clang-23 6-arg form), unused
    g4[0] = 0; g4[1] = 0; g4[2] = 0; g4[3] = 0;
    g4[4] = 0; g4[5] = 0; g4[6] = 0; g4[7] = 0;

    __builtin_amdgcn_tensor_load_to_lds(g0, g1, g2, g3, g4, 0);
    __builtin_amdgcn_s_wait_tensorcnt(0);
  }
  __syncthreads();

  // repack f32 [s][c][p] -> f16 vmain[s][p][c] (+ zero pads, zero vfft)
  for (int i = tid; i < NSAMP * 2 * 160; i += NTHREADS) {
    int s = i / 320;
    int r = i - s * 320;
    int p = r / 160;
    int c = r - p * 160;
    float v = (c < NCH) ? stage[(s * NCH + c) * 2 + p] : 0.f;
    vmain[s][p][c] = (_Float16)v;
    vfft [s][p][c] = (_Float16)0.f;
  }
  __syncthreads();

  // -- stage 2: 143-point DFT magnitude as WMMA GEMM (M=144,K=160,N=16) ------
  for (int tile = wave; tile < 9; tile += 8) {
    v8f re = {}, im = {};
    const int col = ln16, s = col >> 1, p = col & 1;
#pragma unroll
    for (int ks = 0; ks < 5; ++ks) {
      v16h ac = *(const v16h*)(tw + (size_t)(( 0 + tile * 5 + ks) * 32 + lane) * 16);
      v16h as = *(const v16h*)(tw + (size_t)((45 + tile * 5 + ks) * 32 + lane) * 16);
      v16h bf;
#pragma unroll
      for (int e = 0; e < 16; ++e) bf[e] = vmain[s][p][ks * 32 + kmapB(e, g)];
      re = __builtin_amdgcn_wmma_f32_16x16x32_f16(false, ac, false, bf, (short)0, re, false, false);
      im = __builtin_amdgcn_wmma_f32_16x16x32_f16(false, as, false, bf, (short)0, im, false, false);
    }
#pragma unroll
    for (int r = 0; r < 8; ++r) {
      int k = tile * 16 + r + g * 8;
      if (k < NCH)
        vfft[s][p][k] = (_Float16)sqrtf(re[r] * re[r] + im[r] * im[r]);
    }
  }
  __syncthreads();

  // -- stage 3: main conv chain  2->16->28->28->28 ---------------------------
  conv_valu(&vmain[0][0][0], 320, 160, 2, 16, 71, w1, bufA, tid);
  __syncthreads();
  ln_relu(bufA, 16, 71, g1, b1, tid);
  __syncthreads();
  conv_wmma<16, 71, 28, 35>(bufA, wt + WT2_OFF, bufB, tid);
  __syncthreads();
  ln_relu(bufB, 28, 35, g2, b2, tid);
  __syncthreads();
  conv_wmma<28, 35, 28, 17>(bufB, wt + WT3_OFF, bufA, tid);
  __syncthreads();
  ln_relu(bufA, 28, 17, g3, b3, tid);
  __syncthreads();
  conv_wmma<28, 17, 28, 8>(bufA, wt + WT4_OFF, bufB, tid);
  __syncthreads();
  ln_relu(bufB, 28, 8, g4, b4, tid);
  __syncthreads();
  // store: out[b][kind*28+oc][s][t]
  for (int i = tid; i < NSAMP * 28 * 8; i += NTHREADS) {
    int s  = i / 224;
    int r  = i - s * 224;
    int oc = r / 8;
    int t  = r - oc * 8;
    out[b * 4096 + (kind * 28 + oc) * 64 + s * 8 + t] = (float)bufB[s * SSTRIDE + oc * 8 + t];
  }
  __syncthreads();

  // -- stage 4: FFT-branch chain  2->4->4->4->4 (tiny, VALU) -----------------
  conv_valu(&vfft[0][0][0], 320, 160, 2, 4, 71, f1, bufA, tid);
  __syncthreads();
  ln_relu(bufA, 4, 71, g1, b1, tid);
  __syncthreads();
  conv_valu(bufA, SSTRIDE, 71, 4, 4, 35, f2, bufB, tid);
  __syncthreads();
  ln_relu(bufB, 4, 35, g2, b2, tid);
  __syncthreads();
  conv_valu(bufB, SSTRIDE, 35, 4, 4, 17, f3, bufA, tid);
  __syncthreads();
  ln_relu(bufA, 4, 17, g3, b3, tid);
  __syncthreads();
  conv_valu(bufA, SSTRIDE, 17, 4, 4, 8, f4, bufB, tid);
  __syncthreads();
  ln_relu(bufB, 4, 8, g4, b4, tid);
  __syncthreads();
  // store: out[b][56 + kind*4 + oc][s][t]
  for (int i = tid; i < NSAMP * 4 * 8; i += NTHREADS) {
    int s  = i / 32;
    int r  = i - s * 32;
    int oc = r / 8;
    int t  = r - oc * 8;
    out[b * 4096 + (56 + kind * 4 + oc) * 64 + s * 8 + t] = (float)bufB[s * SSTRIDE + oc * 8 + t];
  }
}

extern "C" void kernel_launch(void* const* d_in, const int* in_sizes, int n_in,
                              void* d_out, int out_size, void* d_ws, size_t ws_size,
                              hipStream_t stream) {
  (void)n_in; (void)out_size; (void)ws_size;
  const float* x  = (const float*)d_in[0];
  const float* w1 = (const float*)d_in[1];
  const float* w2 = (const float*)d_in[2];
  const float* w3 = (const float*)d_in[3];
  const float* w4 = (const float*)d_in[4];
  const float* f1 = (const float*)d_in[5];
  const float* f2 = (const float*)d_in[6];
  const float* f3 = (const float*)d_in[7];
  const float* f4 = (const float*)d_in[8];
  const float* g1 = (const float*)d_in[9];
  const float* b1 = (const float*)d_in[10];
  const float* g2 = (const float*)d_in[11];
  const float* b2 = (const float*)d_in[12];
  const float* g3 = (const float*)d_in[13];
  const float* b3 = (const float*)d_in[14];
  const float* g4 = (const float*)d_in[15];
  const float* b4 = (const float*)d_in[16];

  _Float16* tw = (_Float16*)d_ws;            // 90 KiB twiddle fragments
  _Float16* wt = tw + TW_HALFS;              // 16 KiB packed conv weight fragments

  dft_twiddle_init<<<dim3(90), dim3(32), 0, stream>>>(tw);
  weight_pack_init<<<dim3(16), dim3(32), 0, stream>>>(w2, w3, w4, wt);

  int nb = in_sizes[0] / (NCH * XCSTRIDE);   // batch count (4096 for reference shapes)
  dim3 grid((unsigned)nb, 2);
  spectral_pipeline<<<grid, dim3(NTHREADS), 0, stream>>>(
      x, w1, f1, f2, f3, f4,
      g1, b1, g2, b2, g3, b3, g4, b4, tw, wt, (float*)d_out);
}
